// GaussianGraph_40046275068588
// MI455X (gfx1250) — matrix operand order repaired and checked by reading, not compiled, not run on hardware
//
#include <hip/hip_runtime.h>
#include <hip/hip_bf16.h>
#include <math.h>

// Problem constants (from reference)
#define BB    4
#define VV    8
#define HH    128
#define WW    128
#define CC    64
#define VP    7          // V - WS
#define HWPX  16384      // H*W
#define NIMG  32         // B*V
#define NKC   18         // K = 9*64 = 576 -> 18 chunks of 32
#define FRAG_ELEMS 36864 // per conv: 18 kchunks * 4 ntiles * 32 lanes * 16 bf16
#define GAMMA_F 0.1f
#define EPS_F   1e-8f
// Padded (halo) activation layout: 1-pixel zero border, pitch 130
#define HP    130
#define WP    130
#define NHALO 516        // border pixels per image: 2*130 + 2*128

typedef __attribute__((ext_vector_type(16))) __bf16 v16bf;
typedef __attribute__((ext_vector_type(8)))  float  v8f;

union AF { uint4 q[2]; v16bf v; };

static __device__ __forceinline__ unsigned short f2bf(float f) {
  // round-to-nearest-even f32 -> bf16 (bit manipulation, no __bf16 arithmetic)
  unsigned u = __builtin_bit_cast(unsigned, f);
  unsigned r = u + 0x7FFFu + ((u >> 16) & 1u);
  return (unsigned short)(r >> 16);
}

// ---------------------------------------------------------------------------
// Kernel 1: invert 28 affine extrinsics (adjugate 3x3), zero mask counters
// ---------------------------------------------------------------------------
__global__ void prep_kernel(const float* __restrict__ extr,
                            float* __restrict__ einv, int* __restrict__ cnt) {
  int i = threadIdx.x;
  if (i >= BB * VP) return;
  cnt[i] = 0;
  int b = i / VP, j = i % VP, k = j + 1;
  const float* E = extr + (size_t)(b * VV + k) * 16;
  float r00=E[0], r01=E[1], r02=E[2],  t0=E[3];
  float r10=E[4], r11=E[5], r12=E[6],  t1=E[7];
  float r20=E[8], r21=E[9], r22=E[10], t2=E[11];
  float c00 =  (r11*r22 - r12*r21);
  float c01 = -(r10*r22 - r12*r20);
  float c02 =  (r10*r21 - r11*r20);
  float id  = 1.0f / (r00*c00 + r01*c01 + r02*c02);
  float i00 = c00*id;
  float i01 = -(r01*r22 - r02*r21)*id;
  float i02 =  (r01*r12 - r02*r11)*id;
  float i10 = c01*id;
  float i11 =  (r00*r22 - r02*r20)*id;
  float i12 = -(r00*r12 - r02*r10)*id;
  float i20 = c02*id;
  float i21 = -(r00*r21 - r01*r20)*id;
  float i22 =  (r00*r11 - r01*r10)*id;
  float* o = einv + i * 12;
  o[0]=i00; o[1]=i01; o[2]=i02;  o[3] = -(i00*t0 + i01*t1 + i02*t2);
  o[4]=i10; o[5]=i11; o[6]=i12;  o[7] = -(i10*t0 + i11*t1 + i12*t2);
  o[8]=i20; o[9]=i21; o[10]=i22; o[11]= -(i20*t0 + i21*t1 + i22*t2);
}

// ---------------------------------------------------------------------------
// Kernel 2: project points, store gather index (or -1), count valid mask
// ---------------------------------------------------------------------------
__global__ __launch_bounds__(256)
void count_kernel(const float* __restrict__ means,
                  const float* __restrict__ intr,
                  const float* __restrict__ einv,
                  int* __restrict__ idxArr, int* __restrict__ cnt) {
  int i = blockIdx.x * 256 + threadIdx.x;            // 0 .. B*VP*HWPX-1
  int n  = i & (HWPX - 1);
  int jj = i >> 14;                                  // b*VP + j
  int b = jj / VP, j = jj % VP, k = j + 1;
  const float* P = means + ((size_t)(b * VV + j) * HWPX + n) * 3;
  float x = P[0], y = P[1], z = P[2];
  const float* M = einv + (size_t)jj * 12;
  float cx = M[0]*x + M[1]*y + M[2]*z  + M[3];
  float cy = M[4]*x + M[5]*y + M[6]*z  + M[7];
  float cz = M[8]*x + M[9]*y + M[10]*z + M[11];
  const float* K = intr + (size_t)(b * VV + k) * 9;
  float px_ = K[0]*cx + K[1]*cy + K[2]*cz;
  float py_ = K[3]*cx + K[4]*cy + K[5]*cz;
  float pz_ = K[6]*cx + K[7]*cy + K[8]*cz;
  float iz  = 1.0f / (pz_ + EPS_F);
  float nx = px_ * iz, ny = py_ * iz;
  bool valid = (cz > EPS_F) && (nx >= 0.0f) && (nx < 1.0f)
                            && (ny >= 0.0f) && (ny < 1.0f);
  int px = (int)floorf(nx * (float)WW);  px = min(max(px, 0), WW - 1);
  int py = (int)floorf(ny * (float)HH);  py = min(max(py, 0), HH - 1);
  idxArr[i] = valid ? (py * WW + px) : -1;
  if (valid) atomicAdd(&cnt[jj], 1);
}

// ---------------------------------------------------------------------------
// Kernel 3: zero the 1-pixel halo border of both padded activation buffers
// ---------------------------------------------------------------------------
__global__ __launch_bounds__(256)
void halo_kernel(unsigned short* __restrict__ featsbf,
                 unsigned short* __restrict__ x1bf) {
  int i = blockIdx.x * 256 + threadIdx.x;            // 0 .. 2*NIMG*NHALO-1
  int buf = i / (NIMG * NHALO);
  int r   = i % (NIMG * NHALO);
  int img = r / NHALO;
  int e   = r % NHALO;
  int h, w;
  if (e < 130)      { h = 0;           w = e; }
  else if (e < 260) { h = HP - 1;      w = e - 130; }
  else if (e < 388) { h = e - 260 + 1; w = 0; }
  else              { h = e - 388 + 1; w = WP - 1; }
  unsigned short* base =
      (buf ? x1bf : featsbf) + (((size_t)img * HP + h) * WP + w) * CC;
  uint4 z = make_uint4(0, 0, 0, 0);
  uint4* d = (uint4*)base;
#pragma unroll
  for (int q = 0; q < 8; ++q) d[q] = z;
}

// ---------------------------------------------------------------------------
// Kernel 4: blend features, emit bf16 conv input into padded NHWC buffer
// ---------------------------------------------------------------------------
__global__ __launch_bounds__(256)
void fuse_kernel(const float* __restrict__ gs,
                 const int* __restrict__ idxArr, const int* __restrict__ cnt,
                 unsigned short* __restrict__ featsbf) {
  int p = blockIdx.x * 256 + threadIdx.x;            // 0 .. NIMG*HWPX-1
  int n = p & (HWPX - 1);
  int img = p >> 14;                                 // b*VV + v
  int b = img >> 3, v = img & 7;
  int h = n >> 7, w = n & 127;
  const float4* aj = (const float4*)(gs + ((size_t)img * HWPX + n) * CC);
  const float4* ak = nullptr;
  float wgt = 0.0f;
  if (v < VP) {
    int li = (b * VP + v) * HWPX + n;
    int idx = idxArr[li];
    wgt = (GAMMA_F / (float)HWPX) * (float)cnt[b * VP + v];
    if (idx >= 0)
      ak = (const float4*)(gs + (((size_t)img + 1) * HWPX + (size_t)idx) * CC);
  }
  float inv = 1.0f / (1.0f + wgt);
  uint4* dst = (uint4*)(featsbf +
                        (((size_t)img * HP + (h + 1)) * WP + (w + 1)) * CC);
#pragma unroll
  for (int c8 = 0; c8 < 8; ++c8) {
    float4 a0 = aj[2*c8], a1 = aj[2*c8 + 1];
    float4 u0 = make_float4(0,0,0,0), u1 = make_float4(0,0,0,0);
    if (ak) { u0 = ak[2*c8]; u1 = ak[2*c8 + 1]; }
    float r0=(a0.x+u0.x*wgt)*inv, r1=(a0.y+u0.y*wgt)*inv;
    float r2=(a0.z+u0.z*wgt)*inv, r3=(a0.w+u0.w*wgt)*inv;
    float r4=(a1.x+u1.x*wgt)*inv, r5=(a1.y+u1.y*wgt)*inv;
    float r6=(a1.z+u1.z*wgt)*inv, r7=(a1.w+u1.w*wgt)*inv;
    uint4 o;
    o.x = (unsigned)f2bf(r0) | ((unsigned)f2bf(r1) << 16);
    o.y = (unsigned)f2bf(r2) | ((unsigned)f2bf(r3) << 16);
    o.z = (unsigned)f2bf(r4) | ((unsigned)f2bf(r5) << 16);
    o.w = (unsigned)f2bf(r6) | ((unsigned)f2bf(r7) << 16);
    dst[c8] = o;
  }
}

// ---------------------------------------------------------------------------
// Kernel 5: swizzle OIHW f32 weights into WMMA bf16 B-fragment layout.
// B (32x16, 16-bit): lane half (lane/16) holds Klocal = 8*half + j (j<8), 16 + 8*half + (j-8) (j>=8),
// column n = lane%16. Fragment stored so each lane reads 32 contiguous bytes.
// ---------------------------------------------------------------------------
__global__ __launch_bounds__(256)
void wprep_kernel(const float* __restrict__ w1, const float* __restrict__ w2,
                  unsigned short* __restrict__ frag) {
  int f = blockIdx.x * 256 + threadIdx.x;            // 0 .. 2*FRAG_ELEMS-1
  int conv = f / FRAG_ELEMS;
  int r = f % FRAG_ELEMS;
  int j    = r & 15;
  int lane = (r >> 4) & 31;
  int nt   = (r >> 9) & 3;
  int kc   = r >> 11;
  int Kg = kc * 32 + 8 * (lane >> 4) + (j & 7) + ((j >> 3) << 4); // global K index (0..575)
  int co = nt * 16 + (lane & 15);                    // output channel
  int tap = Kg >> 6;                                 // 0..8 (kh*3+kw)
  int ci  = Kg & 63;                                 // input channel
  const float* wsrc = conv ? w2 : w1;
  float val = wsrc[((size_t)co * CC + ci) * 9 + tap];
  frag[f] = f2bf(val);
}

// ---------------------------------------------------------------------------
// Kernel 6/7: implicit-GEMM 3x3 conv via V_WMMA_F32_16X16X32_BF16.
// One WG (8 waves) per 2 image rows; each wave owns a 32-pixel x 64-channel
// output tile (8 f32 accumulators). B-fragments staged in LDS, each ds-load
// feeds TWO WMMAs (both M halves). The zero halo makes every A-load
// unconditional: one base pointer per lane + constant immediate offsets.
// ---------------------------------------------------------------------------
template <bool GELU_BF16OUT>
__global__ __launch_bounds__(256)
void conv_kernel(const unsigned short* __restrict__ in,
                 const unsigned short* __restrict__ fragG,
                 const float* __restrict__ bias, void* __restrict__ outp) {
  extern __shared__ uint4 sB[];                      // 4608 uint4 = 73728 B
  const uint4* g = (const uint4*)fragG;
  for (int t = threadIdx.x; t < FRAG_ELEMS / 8; t += 256) sB[t] = g[t];
  __syncthreads();

  int img  = blockIdx.x >> 6;
  int hp   = blockIdx.x & 63;                        // row pair
  int wave = threadIdx.x >> 5;
  int lane = threadIdx.x & 31;
  int half = lane >> 4;
  int mn   = lane & 15;
  int h    = hp * 2 + (wave >> 2);                   // this wave's output row
  int w0   = (wave & 3) * 32;                        // 32-pixel strip

  // Per-lane base pointer into the padded buffer (pixel (h, w0+mn)).
  const unsigned short* bp =
      in + (((size_t)img * HP + (h + 1)) * WP + (w0 + mn + 1)) * CC + 8 * half;

  // Warm WGP$/L2 for the three input rows this wave consumes (gfx1250
  // global_prefetch_b8; one 128B pixel per lane covers the strip).
#pragma unroll
  for (int r = -1; r <= 1; ++r) {
    const unsigned short* pp =
        in + (((size_t)img * HP + (h + r + 1)) * WP + (w0 + lane + 1)) * CC;
    __builtin_prefetch((const void*)pp, 0, 3);
  }

  v8f acc[8];
#pragma unroll
  for (int t = 0; t < 8; ++t)
#pragma unroll
    for (int e = 0; e < 8; ++e) acc[t][e] = 0.0f;

#pragma unroll
  for (int kc = 0; kc < NKC; ++kc) {
    int tap = kc >> 1;
    int choff = (kc & 1) << 5;
    int dy = tap / 3 - 1, dx = tap % 3 - 1;
    // Constant byte offset per unrolled chunk -> immediate-offset b128 loads.
    const uint4* p = (const uint4*)(bp + ((dy * WP + dx) * CC + choff));
    AF af0, af1;
    af0.q[0] = p[0];     // M-tile 0, K local 8*half .. +7
    af0.q[1] = p[2];     // M-tile 0, K local 16+8*half .. +7
    af1.q[0] = p[128];   // M-tile 1 = +16 pixels = +2048 B
    af1.q[1] = p[130];
#pragma unroll
    for (int nt = 0; nt < 4; ++nt) {
      AF bf;
      const uint4* sp = &sB[(((size_t)kc * 4 + nt) * 32 + lane) * 2];
      bf.q[0] = sp[0];
      bf.q[1] = sp[1];
      acc[nt] = __builtin_amdgcn_wmma_f32_16x16x32_bf16(
          false, af0.v, false, bf.v, (short)0, acc[nt], false, false);
      acc[4 + nt] = __builtin_amdgcn_wmma_f32_16x16x32_bf16(
          false, af1.v, false, bf.v, (short)0, acc[4 + nt], false, false);
    }
  }

  // D layout: lane l, element e -> pixel row = e + 8*half, column = l%16
#pragma unroll
  for (int t = 0; t < 2; ++t) {
#pragma unroll
    for (int nt = 0; nt < 4; ++nt) {
      int co = nt * 16 + mn;
      float bv = bias[co];
#pragma unroll
      for (int e = 0; e < 8; ++e) {
        int wp = w0 + t * 16 + e + 8 * half;
        float x = acc[t * 4 + nt][e] + bv;
        if (GELU_BF16OUT) {
          // conv1 -> GELU -> bf16 into padded intermediate buffer
          size_t oidx = (((size_t)img * HP + (h + 1)) * WP + (wp + 1)) * CC + co;
          x = 0.5f * x * (1.0f + erff(x * 0.70710678118654752f));
          ((unsigned short*)outp)[oidx] = f2bf(x);
        } else {
          // conv2 -> f32 into unpadded output
          size_t oidx = (((size_t)img * HH + h) * WW + wp) * CC + co;
          ((float*)outp)[oidx] = x;
        }
      }
    }
  }
}

// ---------------------------------------------------------------------------
extern "C" void kernel_launch(void* const* d_in, const int* in_sizes, int n_in,
                              void* d_out, int out_size, void* d_ws, size_t ws_size,
                              hipStream_t stream) {
  (void)in_sizes; (void)n_in; (void)out_size; (void)ws_size;
  const float* means = (const float*)d_in[0];
  // d_in[1] = depths (unused by reference)
  const float* gs    = (const float*)d_in[2];
  const float* intr  = (const float*)d_in[3];
  const float* extr  = (const float*)d_in[4];
  const float* w1    = (const float*)d_in[5];
  const float* b1    = (const float*)d_in[6];
  const float* w2    = (const float*)d_in[7];
  const float* b2    = (const float*)d_in[8];

  // Workspace layout (256B aligned blocks)
  unsigned char* ws = (unsigned char*)d_ws;
  size_t off = 0;
  auto take = [&](size_t bytes) -> unsigned char* {
    unsigned char* p = ws + off;
    off = (off + bytes + 255) & ~(size_t)255;
    return p;
  };
  const size_t padElems = (size_t)NIMG * HP * WP * CC;
  float*          einv    = (float*)take(BB * VP * 12 * sizeof(float));
  int*            cnt     = (int*)take(BB * VP * sizeof(int));
  int*            idxArr  = (int*)take((size_t)BB * VP * HWPX * sizeof(int));
  unsigned short* featsbf = (unsigned short*)take(padElems * 2);
  unsigned short* x1bf    = (unsigned short*)take(padElems * 2);
  unsigned short* frag    = (unsigned short*)take((size_t)2 * FRAG_ELEMS * 2);
  unsigned short* frag1 = frag;
  unsigned short* frag2 = frag + FRAG_ELEMS;

  prep_kernel<<<1, 32, 0, stream>>>(extr, einv, cnt);
  count_kernel<<<(BB * VP * HWPX) / 256, 256, 0, stream>>>(means, intr, einv,
                                                           idxArr, cnt);
  halo_kernel<<<(2 * NIMG * NHALO) / 256, 256, 0, stream>>>(featsbf, x1bf);
  fuse_kernel<<<(NIMG * HWPX) / 256, 256, 0, stream>>>(gs, idxArr, cnt, featsbf);
  wprep_kernel<<<(2 * FRAG_ELEMS) / 256, 256, 0, stream>>>(w1, w2, frag);

  const size_t ldsBytes = (FRAG_ELEMS / 8) * sizeof(uint4); // 73728 B
  conv_kernel<true ><<<(NIMG * HH) / 2, 256, ldsBytes, stream>>>(featsbf, frag1,
                                                                 b1, x1bf);
  conv_kernel<false><<<(NIMG * HH) / 2, 256, ldsBytes, stream>>>(x1bf, frag2,
                                                                 b2, d_out);
}
